// BipartiteGATLayer_82806969466910
// MI455X (gfx1250) — matrix operand order repaired and checked by reading, not compiled, not run on hardware
//
#include <hip/hip_runtime.h>
#include <hip/hip_bf16.h>
#include <math.h>

#define NN    8192
#define INF_  512
#define OUTF  64
#define NBROW 80   // 64 h-columns + 1 e-column + 15 zero pad (5 N-tiles of 16)
#define JPARTS 4
#define JCHUNK (NN / JPARTS)   // 2048

typedef __attribute__((ext_vector_type(16))) _Float16 v16h;
typedef __attribute__((ext_vector_type(8)))  float    v8f;
typedef __attribute__((ext_vector_type(4)))  int      v4i;
typedef __attribute__((ext_vector_type(4)))  float    v4f;

// ---------- K0: W (512x64 f32, row-major) -> Wt (64x512 f16, row-major == W^T)
__global__ void k0_wt(const float* __restrict__ W, _Float16* __restrict__ Wt) {
    int idx = blockIdx.x * blockDim.x + threadIdx.x;   // 64*512 threads
    int n = idx >> 9;          // /512
    int k = idx & 511;
    Wt[n * INF_ + k] = (_Float16)W[k * OUTF + n];
}

// ---------- K1: H = X @ W  (8192x512 @ 512x64), WMMA f16 -> f32
__global__ void k1_h(const float* __restrict__ X, const _Float16* __restrict__ Wt,
                     float* __restrict__ H) {
    int wave  = (blockIdx.x * blockDim.x + threadIdx.x) >> 5;  // 2048 waves
    int lane  = threadIdx.x & 31;
    int mtile = wave >> 2;            // 0..511
    int ntile = wave & 3;             // 0..3
    int mrow  = mtile * 16 + (lane & 15);
    int ncol  = ntile * 16 + (lane & 15);
    int kA0   = (lane < 16) ? 0 : 16;    // A: contiguous K halves (matches B)
    int kB0   = (lane < 16) ? 0 : 16;    // B: contiguous K halves (matches A)
    const float*    xrow = X  + (size_t)mrow * INF_;
    const _Float16* wrow = Wt + (size_t)ncol * INF_;
    v8f acc = {};
    for (int k = 0; k < INF_; k += 32) {
        v4f a0 = *(const v4f*)(xrow + k + kA0);
        v4f a1 = *(const v4f*)(xrow + k + kA0 + 4);
        v4f a2 = *(const v4f*)(xrow + k + kA0 + 8);
        v4f a3 = *(const v4f*)(xrow + k + kA0 + 12);
        v16h a;
#pragma unroll
        for (int e = 0; e < 4; ++e) {
            a[e]      = (_Float16)a0[e];
            a[4 + e]  = (_Float16)a1[e];
            a[8 + e]  = (_Float16)a2[e];
            a[12 + e] = (_Float16)a3[e];
        }
        v16h b = *(const v16h*)(wrow + k + kB0);   // 32B contiguous
        acc = __builtin_amdgcn_wmma_f32_16x16x32_f16(false, a, false, b,
                                                     (short)0, acc, false, false);
    }
    int mbase = mtile * 16 + ((lane < 16) ? 0 : 8);
#pragma unroll
    for (int v = 0; v < 8; ++v)
        H[(size_t)(mbase + v) * OUTF + ncol] = acc[v];
}

// ---------- K2: r[i] = h[i,:] . a_right   (a_right = attention_vector[64:128])
__global__ void k2_r(const float* __restrict__ H, const float* __restrict__ Avec,
                     float* __restrict__ R) {
    int i = blockIdx.x * blockDim.x + threadIdx.x;
    float s = 0.f;
#pragma unroll 8
    for (int k = 0; k < OUTF; ++k) s += H[(size_t)i * OUTF + k] * Avec[OUTF + k];
    R[i] = s;
}

// ---------- K3: rmax = max(r)
__global__ void k3_max(const float* __restrict__ R, float* __restrict__ Rmax) {
    __shared__ float sm[256];
    float m = -__builtin_inff();
    for (int i = threadIdx.x; i < NN; i += 256) m = fmaxf(m, R[i]);
    sm[threadIdx.x] = m;
    __syncthreads();
    for (int s = 128; s > 0; s >>= 1) {
        if (threadIdx.x < (unsigned)s)
            sm[threadIdx.x] = fmaxf(sm[threadIdx.x], sm[threadIdx.x + s]);
        __syncthreads();
    }
    if (threadIdx.x == 0) *Rmax = sm[0];
}

// ---------- K4: Hb (NBROW x 8192 f16, row-major) = [ (e[j]*h[j,n])^T ; e ; 0 ]
__global__ void k4_hb(const float* __restrict__ H, const float* __restrict__ R,
                      const float* __restrict__ Rmax, _Float16* __restrict__ Hb) {
    int j = blockIdx.x * blockDim.x + threadIdx.x;
    float e = __expf(R[j] - *Rmax);
#pragma unroll 8
    for (int n = 0; n < OUTF; ++n)
        Hb[(size_t)n * NN + j] = (_Float16)(e * H[(size_t)j * OUTF + n]);
    Hb[(size_t)OUTF * NN + j] = (_Float16)e;
#pragma unroll
    for (int n = OUTF + 1; n < NBROW; ++n)
        Hb[(size_t)n * NN + j] = (_Float16)0.0f;
}

// 0/1 int pair -> packed f16 pair (0x3C00 = 1.0h); no cvt chains needed.
__device__ __forceinline__ unsigned pack01(int lo, int hi) {
    return ((unsigned)lo + ((unsigned)hi << 16)) * 0x3C00u;
}

// ---------- K5: fused partial  [Num | Den] += Adj(0/1) @ Hb^T  over one j-part
// 2048 waves: itile = w % 512, jpart = w / 512. Deterministic (no atomics).
__global__ void k5_agg(const int* __restrict__ Adj, const _Float16* __restrict__ Hb,
                       float* __restrict__ NumP, float* __restrict__ DenP) {
    int w    = (blockIdx.x * blockDim.x + threadIdx.x) >> 5;
    int lane = threadIdx.x & 31;
    int itile = w & 511;
    int jpart = w >> 9;
    int arowi = itile * 16 + (lane & 15);
    int kA0   = (lane < 16) ? 0 : 16;
    int kB0   = (lane < 16) ? 0 : 16;
    int bn    = lane & 15;
    const int* arow = Adj + (size_t)arowi * NN;
    const _Float16* b0p = Hb + (size_t)(0 * 16 + bn) * NN;
    const _Float16* b1p = Hb + (size_t)(1 * 16 + bn) * NN;
    const _Float16* b2p = Hb + (size_t)(2 * 16 + bn) * NN;
    const _Float16* b3p = Hb + (size_t)(3 * 16 + bn) * NN;
    const _Float16* b4p = Hb + (size_t)(4 * 16 + bn) * NN;
    v8f acc0 = {}, acc1 = {}, acc2 = {}, acc3 = {}, acc4 = {};
    int j0 = jpart * JCHUNK;
    for (int j = j0; j < j0 + JCHUNK; j += 32) {
        __builtin_prefetch(arow + j + 512, 0, 0);   // stream adjacency ahead
        v4i a0 = *(const v4i*)(arow + j + kA0);
        v4i a1 = *(const v4i*)(arow + j + kA0 + 4);
        v4i a2 = *(const v4i*)(arow + j + kA0 + 8);
        v4i a3 = *(const v4i*)(arow + j + kA0 + 12);
        union { unsigned u[8]; v16h h; } au;
        au.u[0] = pack01(a0[0], a0[1]);
        au.u[1] = pack01(a0[2], a0[3]);
        au.u[2] = pack01(a1[0], a1[1]);
        au.u[3] = pack01(a1[2], a1[3]);
        au.u[4] = pack01(a2[0], a2[1]);
        au.u[5] = pack01(a2[2], a2[3]);
        au.u[6] = pack01(a3[0], a3[1]);
        au.u[7] = pack01(a3[2], a3[3]);
        v16h a = au.h;
        v16h b0 = *(const v16h*)(b0p + j + kB0);
        v16h b1 = *(const v16h*)(b1p + j + kB0);
        v16h b2 = *(const v16h*)(b2p + j + kB0);
        v16h b3 = *(const v16h*)(b3p + j + kB0);
        v16h b4 = *(const v16h*)(b4p + j + kB0);
        acc0 = __builtin_amdgcn_wmma_f32_16x16x32_f16(false, a, false, b0, (short)0, acc0, false, false);
        acc1 = __builtin_amdgcn_wmma_f32_16x16x32_f16(false, a, false, b1, (short)0, acc1, false, false);
        acc2 = __builtin_amdgcn_wmma_f32_16x16x32_f16(false, a, false, b2, (short)0, acc2, false, false);
        acc3 = __builtin_amdgcn_wmma_f32_16x16x32_f16(false, a, false, b3, (short)0, acc3, false, false);
        acc4 = __builtin_amdgcn_wmma_f32_16x16x32_f16(false, a, false, b4, (short)0, acc4, false, false);
    }
    float* Num = NumP + (size_t)jpart * NN * OUTF;
    float* Den = DenP + (size_t)jpart * NN;
    int mbase = itile * 16 + ((lane < 16) ? 0 : 8);
#pragma unroll
    for (int v = 0; v < 8; ++v) {
        Num[(size_t)(mbase + v) * OUTF +  0 + bn] = acc0[v];
        Num[(size_t)(mbase + v) * OUTF + 16 + bn] = acc1[v];
        Num[(size_t)(mbase + v) * OUTF + 32 + bn] = acc2[v];
        Num[(size_t)(mbase + v) * OUTF + 48 + bn] = acc3[v];
    }
    if (bn == 0) {
#pragma unroll
        for (int v = 0; v < 8; ++v) Den[mbase + v] = acc4[v];   // column 64 = A@e
    }
}

// ---------- K6: out = elu( (sum_p NumP) / (sum_p DenP) )
__global__ void k6_out(const float* __restrict__ NumP, const float* __restrict__ DenP,
                       float* __restrict__ Out) {
    int idx = blockIdx.x * blockDim.x + threadIdx.x;   // 8192*64
    int i = idx >> 6;   // /64
    float n = 0.f, d = 0.f;
#pragma unroll
    for (int p = 0; p < JPARTS; ++p) {
        n += NumP[(size_t)p * NN * OUTF + idx];
        d += DenP[(size_t)p * NN + i];
    }
    float x = n / d;
    Out[idx] = (x > 0.f) ? x : expm1f(x);
}

extern "C" void kernel_launch(void* const* d_in, const int* in_sizes, int n_in,
                              void* d_out, int out_size, void* d_ws, size_t ws_size,
                              hipStream_t stream) {
    const float* X    = (const float*)d_in[0];   // node_features 8192x512
    const int*   Adj  = (const int*)  d_in[1];   // adjacency 8192x8192 int32
    const float* W    = (const float*)d_in[2];   // weight 512x64
    const float* Avec = (const float*)d_in[3];   // attention_vector 128x1
    float* Out = (float*)d_out;                  // 8192x64 f32

    char* ws = (char*)d_ws;
    float*    H    = (float*)   (ws + 0);          // 2,097,152 B
    _Float16* Wt   = (_Float16*)(ws + 2097152);    //    65,536 B
    float*    R    = (float*)   (ws + 2162688);    //    32,768 B
    float*    Rmax = (float*)   (ws + 2195456);    //       256 B
    _Float16* Hb   = (_Float16*)(ws + 2195712);    // 1,310,720 B
    float*    NumP = (float*)   (ws + 3506432);    // 4 * 2,097,152 B
    float*    DenP = (float*)   (ws + 11895040);   // 4 *    32,768 B  (end ~12.0 MB)

    k0_wt <<<(OUTF * INF_) / 256, 256, 0, stream>>>(W, Wt);
    k1_h  <<<256, 256, 0, stream>>>(X, Wt, H);            // 2048 waves
    k2_r  <<<NN / 256, 256, 0, stream>>>(H, Avec, R);
    k3_max<<<1, 256, 0, stream>>>(R, Rmax);
    k4_hb <<<NN / 256, 256, 0, stream>>>(H, R, Rmax, Hb);
    k5_agg<<<256, 256, 0, stream>>>(Adj, Hb, NumP, DenP); // 2048 waves
    k6_out<<<(NN * OUTF) / 256, 256, 0, stream>>>(NumP, DenP, Out);
}
